// SelfAttentionModule_38336878084810
// MI455X (gfx1250) — matrix operand order repaired and checked by reading, not compiled, not run on hardware
//
#include <hip/hip_runtime.h>

// ---------------------------------------------------------------------------
// CDNA5 (gfx1250) self-attention: N=4096, D=1024, fp32 in/out.
// bf16 WMMA (v_wmma_f32_16x16x32_bf16) for all five GEMMs, fp32 accumulate.
// Block = 4 waves arranged 2x2; wave tile = 64x64 (16 WMMA accum tiles).
// ---------------------------------------------------------------------------

typedef __attribute__((ext_vector_type(16))) __bf16 v16bf;
typedef __attribute__((ext_vector_type(8)))  __bf16 v8bf;
typedef __attribute__((ext_vector_type(8)))  float  v8f;

#define ATT_N 4096
#define ATT_D 1024

// ---------------------------------------------------------------------------
// fp32 -> bf16 conversion
// ---------------------------------------------------------------------------
__global__ void attn_cvt_f32_bf16(const float* __restrict__ src,
                                  __bf16* __restrict__ dst, int n) {
  int i = blockIdx.x * blockDim.x + threadIdx.x;
  if (i < n) dst[i] = (__bf16)src[i];
}

// ---------------------------------------------------------------------------
// bf16 transpose (rows x cols) -> (cols x rows), 32x32 LDS tiles
// ---------------------------------------------------------------------------
__global__ void attn_transpose_bf16(const __bf16* __restrict__ in,
                                    __bf16* __restrict__ out,
                                    int rows, int cols) {
  __shared__ __bf16 tile[32][33];
  const int bx = blockIdx.x * 32;  // col base
  const int by = blockIdx.y * 32;  // row base
  const int tx = threadIdx.x;      // 0..31
  const int ty = threadIdx.y;      // 0..7
#pragma unroll
  for (int i = 0; i < 32; i += 8)
    tile[ty + i][tx] = in[(size_t)(by + ty + i) * cols + (bx + tx)];
  __syncthreads();
#pragma unroll
  for (int i = 0; i < 32; i += 8)
    out[(size_t)(bx + ty + i) * rows + (by + tx)] = tile[tx][ty + i];
}

// ---------------------------------------------------------------------------
// A-fragment loader: 16x32 bf16 A tile, per-lane (m = lane&15, half = lane>>4)
//   elems 0..7  : K = half*16 + 0..7       (16B contiguous)
//   elems 8..15 : K = half*16 + 8 + 0..7   (16B contiguous)
// p must already point at &A[row*lda + k0 + half*16].
// ---------------------------------------------------------------------------
__device__ __forceinline__ v16bf attn_load_a(const __bf16* p) {
  v8bf lo = *reinterpret_cast<const v8bf*>(p);
  v8bf hi = *reinterpret_cast<const v8bf*>(p + 8);
  v16bf r;
#pragma unroll
  for (int i = 0; i < 8; ++i) { r[i] = lo[i]; r[i + 8] = hi[i]; }
  return r;
}

// ---------------------------------------------------------------------------
// GEMM: C[M,Ncols] = scale * (A[M,K] x B[K,Ncols]) (+ bias[col])
// B is supplied as S (row-major) where S row j == B column j (contiguous K),
// matching the B-fragment order this code uses (same as A): lane (n, half) reads 16 contiguous
// K-elements at K = half*16.
// Block = 128 threads (4 wave32s) arranged 2x2 -> block tile 128(M) x 128(N).
// Wave tile = 64(M) x 64(N) = 4x4 WMMA tiles, fp32 accumulators (128 VGPRs).
// Per k-step: 4 A-frags + 4 B-frags (8KB/wave) feed 16 WMMAs (262 kFLOP):
// ~32 flop/byte from L1, with waves sharing A rows / B cols pairwise.
// Grid must tile exactly (no bounds checks -> EXEC all ones for WMMA).
// ---------------------------------------------------------------------------
template <bool HAS_BIAS, bool OUT_BF16>
__global__ __launch_bounds__(128)
void attn_gemm_bf16_wmma(const __bf16* __restrict__ A, int lda,
                         const __bf16* __restrict__ S, int ldb,
                         const float* __restrict__ bias,
                         void* __restrict__ Cout, int ldc,
                         int K, float scale) {
  const int lane = threadIdx.x & 31;
  const int wave = threadIdx.x >> 5;
  const int n    = lane & 15;   // row (A) / col (B,C) within 16
  const int half = lane >> 4;

  const int colW = blockIdx.x * 128 + (wave & 1) * 64;
  const int rowW = blockIdx.y * 128 + (wave >> 1) * 64;

  const __bf16* pa = A + (size_t)(rowW + n) * lda + half * 16;
  const __bf16* pb = S + (size_t)(colW + n) * ldb + half * 16;
  const size_t astride = (size_t)16 * lda;
  const size_t bstride = (size_t)16 * ldb;

  v8f acc[4][4] = {};

  for (int k0 = 0; k0 < K; k0 += 32) {
    v16bf a[4];
    v16bf b[4];
#pragma unroll
    for (int i = 0; i < 4; ++i)
      a[i] = attn_load_a(pa + i * astride + k0);
#pragma unroll
    for (int t = 0; t < 4; ++t)
      b[t] = *reinterpret_cast<const v16bf*>(pb + t * bstride + k0);
#pragma unroll
    for (int i = 0; i < 4; ++i) {
#pragma unroll
      for (int t = 0; t < 4; ++t) {
        acc[i][t] = __builtin_amdgcn_wmma_f32_16x16x32_bf16(
            false, a[i], false, b[t], (short)0, acc[i][t], false, false);
      }
    }
  }

  float bv[4];
#pragma unroll
  for (int t = 0; t < 4; ++t)
    bv[t] = HAS_BIAS ? bias[colW + t * 16 + n] : 0.0f;

  // C/D layout: lane (n, half), vgpr r -> C[rowW + i*16 + half*8 + r][colW + t*16 + n]
#pragma unroll
  for (int i = 0; i < 4; ++i) {
#pragma unroll
    for (int t = 0; t < 4; ++t) {
#pragma unroll
      for (int r = 0; r < 8; ++r) {
        const int row = rowW + i * 16 + half * 8 + r;
        const int col = colW + t * 16 + n;
        const float v = acc[i][t][r] * scale + bv[t];
        if (OUT_BF16)
          reinterpret_cast<__bf16*>(Cout)[(size_t)row * ldc + col] = (__bf16)v;
        else
          reinterpret_cast<float*>(Cout)[(size_t)row * ldc + col] = v;
      }
    }
  }
}

// ---------------------------------------------------------------------------
// Row softmax, in place: reads 4096 fp32, writes 4096 bf16 over same storage.
// One 256-thread block per row; each thread owns 16 strided elements.
// ---------------------------------------------------------------------------
__global__ __launch_bounds__(256)
void attn_softmax_row_inplace(float* __restrict__ scores, int ncols) {
  const int row = blockIdx.x;
  float* rp = scores + (size_t)row * ncols;
  __bf16* op = reinterpret_cast<__bf16*>(rp);
  const int tid = threadIdx.x;

  float v[16];
  float mx = -__builtin_inff();
#pragma unroll
  for (int i = 0; i < 16; ++i) {
    v[i] = rp[tid + i * 256];
    mx = fmaxf(mx, v[i]);
  }

  __shared__ float red[256];
  red[tid] = mx;
  __syncthreads();
  for (int s = 128; s > 0; s >>= 1) {
    if (tid < s) red[tid] = fmaxf(red[tid], red[tid + s]);
    __syncthreads();
  }
  mx = red[0];
  __syncthreads();

  float sum = 0.0f;
#pragma unroll
  for (int i = 0; i < 16; ++i) {
    v[i] = __expf(v[i] - mx);
    sum += v[i];
  }
  red[tid] = sum;
  __syncthreads();
  for (int s = 128; s > 0; s >>= 1) {
    if (tid < s) red[tid] += red[tid + s];
    __syncthreads();
  }
  const float inv = 1.0f / red[0];
#pragma unroll
  for (int i = 0; i < 16; ++i)
    op[tid + i * 256] = (__bf16)(v[i] * inv);
}

// ---------------------------------------------------------------------------
// Host launcher
// ---------------------------------------------------------------------------
extern "C" void kernel_launch(void* const* d_in, const int* in_sizes, int n_in,
                              void* d_out, int out_size, void* d_ws, size_t ws_size,
                              hipStream_t stream) {
  (void)in_sizes; (void)n_in; (void)out_size; (void)ws_size;
  const int N = ATT_N, D = ATT_D;

  const float* x  = (const float*)d_in[0];
  const float* Wq = (const float*)d_in[1];
  const float* bq = (const float*)d_in[2];
  const float* Wk = (const float*)d_in[3];
  const float* bk = (const float*)d_in[4];
  const float* Wv = (const float*)d_in[5];
  const float* bv = (const float*)d_in[6];

  char* ws = (char*)d_ws;
  size_t off = 0;
  auto carve = [&](size_t bytes) -> void* {
    void* p = ws + off;
    off = (off + bytes + 255) & ~(size_t)255;
    return p;
  };

  __bf16* xb     = (__bf16*)carve((size_t)N * D * 2);   //  8 MB
  __bf16* Wqb    = (__bf16*)carve((size_t)D * D * 2);   //  2 MB
  __bf16* Wkb    = (__bf16*)carve((size_t)D * D * 2);
  __bf16* Wvb    = (__bf16*)carve((size_t)D * D * 2);
  __bf16* Qb     = (__bf16*)carve((size_t)N * D * 2);   //  8 MB
  __bf16* Kb     = (__bf16*)carve((size_t)N * D * 2);
  __bf16* Vb     = (__bf16*)carve((size_t)N * D * 2);
  __bf16* Vtb    = (__bf16*)carve((size_t)D * N * 2);
  float*  scores = (float*) carve((size_t)N * N * 4);   // 64 MB (attn bf16 reuses this in place)

  // --- fp32 -> bf16 conversions -------------------------------------------
  {
    const int nx = N * D;        // 4M
    attn_cvt_f32_bf16<<<nx / 256, 256, 0, stream>>>(x, xb, nx);
    const int nw = D * D;        // 1M
    attn_cvt_f32_bf16<<<nw / 256, 256, 0, stream>>>(Wq, Wqb, nw);
    attn_cvt_f32_bf16<<<nw / 256, 256, 0, stream>>>(Wk, Wkb, nw);
    attn_cvt_f32_bf16<<<nw / 256, 256, 0, stream>>>(Wv, Wvb, nw);
  }

  // --- projections: Q/K/V = x @ W^T + b  (W rows are B columns) -----------
  {
    dim3 grid(D / 128, N / 128);  // (8, 32)
    dim3 block(128);
    attn_gemm_bf16_wmma<true, true><<<grid, block, 0, stream>>>(
        xb, D, Wqb, D, bq, (void*)Qb, D, D, 1.0f);
    attn_gemm_bf16_wmma<true, true><<<grid, block, 0, stream>>>(
        xb, D, Wkb, D, bk, (void*)Kb, D, D, 1.0f);
    attn_gemm_bf16_wmma<true, true><<<grid, block, 0, stream>>>(
        xb, D, Wvb, D, bv, (void*)Vb, D, D, 1.0f);
  }

  // --- V^T (for contiguous B-fragments in attn @ V) -----------------------
  attn_transpose_bf16<<<dim3(D / 32, N / 32), dim3(32, 8), 0, stream>>>(
      Vb, Vtb, N, D);

  // --- scores = (Q @ K^T) / sqrt(D)  (K rows are B columns) ---------------
  {
    dim3 grid(N / 128, N / 128);  // (32, 32)
    attn_gemm_bf16_wmma<false, false><<<grid, dim3(128), 0, stream>>>(
        Qb, D, Kb, D, nullptr, (void*)scores, N, D, 0.03125f /* 1/sqrt(1024) */);
  }

  // --- softmax rows, in place: fp32 scores -> bf16 attn -------------------
  attn_softmax_row_inplace<<<N, 256, 0, stream>>>(scores, N);

  // --- out = attn @ V  (attn bf16 with element-stride 2N; V^T rows = B cols)
  {
    dim3 grid(D / 128, N / 128);  // (8, 32)
    attn_gemm_bf16_wmma<false, false><<<grid, dim3(128), 0, stream>>>(
        (const __bf16*)scores, 2 * N, Vtb, N, nullptr, d_out, D, N, 1.0f);
  }
}
